// STConv_18176301597614
// MI455X (gfx1250) — compile-verified, matrix operand order checked by reading, not run on hardware
//
#include <hip/hip_runtime.h>
#include <hip/hip_bf16.h>
#include <stdint.h>

typedef _Float16 v16h __attribute__((ext_vector_type(16)));
typedef _Float16 h8v  __attribute__((ext_vector_type(8)));
typedef float    v8f  __attribute__((ext_vector_type(8)));

#define WMMA(a,b,c) __builtin_amdgcn_wmma_f32_16x16x32_f16(false,(a),false,(b),(short)0,(c),false,false)

// ---- problem dims ----
#define Bb    32
#define TW    40
#define NN    50
#define CIN   2
#define HH    128
#define CO    64
#define EE    800
#define T1    38                    // after tconv1
#define T2N   36                    // after tconv2
#define T3    34                    // after conv3
#define S1    (Bb*T1)               // 1216 samples for cheb
static const size_t TXS = (size_t)S1*NN*HH;   // 7,782,400 elements per Tx buffer

// ---------------------------------------------------------------------------
// WMMA f16 fragment load: halves 0..7 at p[0..7], halves 8..15 at p[16..23].
// All call sites are 16-byte aligned -> two global_load_b128 per fragment.
// ---------------------------------------------------------------------------
__device__ __forceinline__ v16h frag_load(const _Float16* p) {
  union { v16h v; h8v h[2]; } u;
  u.h[0] = *(const h8v*)(p);
  u.h[1] = *(const h8v*)(p + 16);
  return u.v;
}

// ---------------------------------------------------------------------------
// Weight prep: K x N row-major f32  ->  N x K row-major f16 (coalesced writes)
// ---------------------------------------------------------------------------
__global__ void transpose_f16_kernel(const float* __restrict__ src, _Float16* __restrict__ dst,
                                     int K, int N) {
  int i = blockIdx.x * blockDim.x + threadIdx.x;
  if (i >= K * N) return;
  int n = i / K, k = i % K;
  dst[i] = (_Float16)src[k * N + n];
}
__global__ void convert_f16_kernel(const float* __restrict__ src, _Float16* __restrict__ dst,
                                   int total) {
  int i = blockIdx.x * blockDim.x + threadIdx.x;
  if (i < total) dst[i] = (_Float16)src[i];
}

// ---------------------------------------------------------------------------
// Kernel 1: temporal conv 1 (GLU): T0h[b,t,n,h] = relu(P*sigmoid(Q)+R), f16 out
// ---------------------------------------------------------------------------
__global__ void tconv1_kernel(const float* __restrict__ X,
                              const float* __restrict__ w1, const float* __restrict__ b1,
                              const float* __restrict__ w2, const float* __restrict__ b2,
                              const float* __restrict__ w3, const float* __restrict__ b3,
                              _Float16* __restrict__ T0h) {
  int idx = blockIdx.x * blockDim.x + threadIdx.x;
  const int total = Bb * T1 * NN * HH;
  if (idx >= total) return;
  int h = idx & (HH - 1);
  int n = (idx >> 7) % NN;
  int t = (idx / (HH * NN)) % T1;
  int b = idx / (HH * NN * T1);
  float P = b1[h], Q = b2[h], R = b3[h];
#pragma unroll
  for (int k = 0; k < 3; ++k) {
#pragma unroll
    for (int c = 0; c < CIN; ++c) {
      float x = X[(((size_t)b * TW + t + k) * NN + n) * CIN + c];
      int wi = (k * CIN + c) * HH + h;
      P = fmaf(x, w1[wi], P);
      Q = fmaf(x, w2[wi], Q);
      R = fmaf(x, w3[wi], R);
    }
  }
  float s = 1.0f / (1.0f + __expf(-Q));
  T0h[idx] = (_Float16)fmaxf(fmaf(P, s, R), 0.0f);
}

// ---------------------------------------------------------------------------
// Kernel 2: per-sample symmetric-normalized edge weights
// ---------------------------------------------------------------------------
__global__ void edgew_kernel(const int* __restrict__ ei, float* __restrict__ wE) {
  __shared__ int   deg[64];
  __shared__ float dis[64];
  int s = blockIdx.x;                 // 0..S1-1
  int b = s / T1, t = s % T1;
  const int* row = ei + ((size_t)(b * TW + t) * 2 + 0) * EE;
  const int* col = row + EE;
  int tid = threadIdx.x;
  if (tid < 64) deg[tid] = 0;
  __syncthreads();
  for (int e = tid; e < EE; e += blockDim.x) atomicAdd(&deg[row[e]], 1);
  __syncthreads();
  if (tid < NN) {
    float d = (float)deg[tid];
    dis[tid] = (d > 0.0f) ? rsqrtf(d) : 0.0f;
  }
  __syncthreads();
  for (int e = tid; e < EE; e += blockDim.x)
    wE[(size_t)s * EE + e] = -dis[row[e]] * dis[col[e]];
}

// ---------------------------------------------------------------------------
// Kernel 3: graph propagation  out = f16(alpha * (L @ in) + beta * other)
// one block per sample; f32 accumulation in LDS (ds_add_f32 atomics)
// ---------------------------------------------------------------------------
__global__ void prop_kernel(const _Float16* __restrict__ in, const _Float16* __restrict__ other,
                            const float* __restrict__ wE, const int* __restrict__ ei,
                            float alpha, float beta, _Float16* __restrict__ out) {
  __shared__ float acc[NN * HH];      // 25.6 KB
  int s = blockIdx.x;
  int b = s / T1, t = s % T1;
  const int* row = ei + ((size_t)(b * TW + t) * 2) * EE;
  const int* col = row + EE;
  const float* w = wE + (size_t)s * EE;
  const _Float16* xi = in + (size_t)s * NN * HH;
  int tid = threadIdx.x;
  for (int i = tid; i < NN * HH; i += blockDim.x) acc[i] = 0.0f;
  __syncthreads();
  for (int i = tid; i < EE * HH; i += blockDim.x) {
    int e = i >> 7, h = i & (HH - 1);
    float v = w[e] * (float)xi[col[e] * HH + h];
    atomicAdd(&acc[row[e] * HH + h], v);
  }
  __syncthreads();
  _Float16* o = out + (size_t)s * NN * HH;
  const _Float16* oth = other + (size_t)s * NN * HH;
  for (int i = tid; i < NN * HH; i += blockDim.x)
    o[i] = (_Float16)fmaf(alpha, acc[i], beta * (float)oth[i]);
}

// ---------------------------------------------------------------------------
// Kernel 4: Chebyshev einsum WMMA GEMM: (60800 x 384) @ (384 x 128)
// A = contiguous f16 {T0h,Tx1h,Tx2h}; B = pre-transposed f16 (128 x 384)
// epilogue: bias + relu -> G (f16)
// ---------------------------------------------------------------------------
__global__ void cheb_gemm_kernel(const _Float16* __restrict__ Txh,
                                 const _Float16* __restrict__ Wt,   // 128 x 384, n-major
                                 const float* __restrict__ bias,
                                 _Float16* __restrict__ G) {
  int wave = (blockIdx.x * blockDim.x + threadIdx.x) >> 5;
  int lane = threadIdx.x & 31;
  int tn = wave & 7;          // 128/16 column tiles
  int tm = wave >> 3;         // 3800 row tiles
  int r  = lane & 15;
  int kb = (lane >> 4) << 3;
  int rowA = tm * 16 + r;
  int colB = tn * 16 + r;
  const _Float16* pb0 = Wt + (size_t)colB * 384;
  v8f acc = {};
  for (int k0 = 0; k0 < 384; k0 += 32) {
    v16h a  = frag_load(Txh + (size_t)(k0 >> 7) * TXS + (size_t)rowA * HH + (k0 & 127) + kb);
    v16h bf = frag_load(pb0 + k0 + kb);
    acc = WMMA(a, bf, acc);
  }
  int n  = lane & 15;
  int mb = (lane >> 4) << 3;
  float bv = bias[tn * 16 + n];
#pragma unroll
  for (int v = 0; v < 8; ++v)
    G[(size_t)(tm * 16 + mb + v) * HH + tn * 16 + n] =
        (_Float16)fmaxf(acc[v] + bv, 0.0f);
}

// ---------------------------------------------------------------------------
// Kernel 5: tconv2 (GLU) triple WMMA GEMM: (57600 x 384) @ (384 x 64) x3
// A gathers G[b][t2+k][n][:] (k-block stride 6400); B pre-transposed (64 x 384)
// f32 output for exact BN statistics
// ---------------------------------------------------------------------------
__global__ void tconv2_gemm_kernel(const _Float16* __restrict__ G,
                                   const _Float16* __restrict__ w1t, const float* __restrict__ b1,
                                   const _Float16* __restrict__ w2t, const float* __restrict__ b2,
                                   const _Float16* __restrict__ w3t, const float* __restrict__ b3,
                                   float* __restrict__ T2) {
  int wave = (blockIdx.x * blockDim.x + threadIdx.x) >> 5;
  int lane = threadIdx.x & 31;
  int tn = wave & 3;          // 64/16
  int tm = wave >> 2;         // 3600
  int r  = lane & 15;
  int kb = (lane >> 4) << 3;
  int rowA = tm * 16 + r;
  int b_  = rowA / (T2N * NN);
  int t2  = (rowA / NN) % T2N;
  int n_  = rowA % NN;
  size_t baseA = ((size_t)(b_ * T1 + t2) * NN + n_) * HH;
  int colB = tn * 16 + r;
  const _Float16* p1 = w1t + (size_t)colB * 384;
  const _Float16* p2 = w2t + (size_t)colB * 384;
  const _Float16* p3 = w3t + (size_t)colB * 384;
  v8f aP = {}, aQ = {}, aR = {};
  for (int k0 = 0; k0 < 384; k0 += 32) {
    v16h a  = frag_load(G + baseA + (size_t)(k0 >> 7) * (NN * HH) + (k0 & 127) + kb);
    v16h f1 = frag_load(p1 + k0 + kb);
    v16h f2 = frag_load(p2 + k0 + kb);
    v16h f3 = frag_load(p3 + k0 + kb);
    aP = WMMA(a, f1, aP);
    aQ = WMMA(a, f2, aQ);
    aR = WMMA(a, f3, aR);
  }
  int n  = lane & 15;
  int mb = (lane >> 4) << 3;
  int co = tn * 16 + n;
  float bb1 = b1[co], bb2 = b2[co], bb3 = b3[co];
#pragma unroll
  for (int v = 0; v < 8; ++v) {
    float P = aP[v] + bb1;
    float Q = aQ[v] + bb2;
    float R = aR[v] + bb3;
    float sg = 1.0f / (1.0f + __expf(-Q));
    T2[(size_t)(tm * 16 + mb + v) * CO + co] = fmaxf(fmaf(P, sg, R), 0.0f);
  }
}

// ---------------------------------------------------------------------------
// Kernel 6: per-node BN statistics (mean + rsqrt(var+eps)) over f32 T2
// ---------------------------------------------------------------------------
__global__ void bnstats_kernel(const float* __restrict__ T2, float* __restrict__ stats) {
  __shared__ float s1[256], s2[256];
  int n = blockIdx.x, tid = threadIdx.x;
  const int per = Bb * T2N * CO;      // 73728
  float a = 0.0f, b = 0.0f;
  for (int i = tid; i < per; i += 256) {
    int bt = i >> 6;                  // b*36 + t2
    int co = i & 63;
    float v = T2[((size_t)bt * NN + n) * CO + co];
    a += v; b = fmaf(v, v, b);
  }
  s1[tid] = a; s2[tid] = b;
  __syncthreads();
  for (int off = 128; off > 0; off >>= 1) {
    if (tid < off) { s1[tid] += s1[tid + off]; s2[tid] += s2[tid + off]; }
    __syncthreads();
  }
  if (tid == 0) {
    float cnt = (float)per;
    float mean = s1[0] / cnt;
    float var  = s2[0] / cnt - mean * mean;
    stats[n]      = mean;
    stats[64 + n] = rsqrtf(var + 1e-5f);
  }
}

// Kernel 7: BN apply: f32 in -> normalized f16 out (feeds conv3 WMMA)
__global__ void bnapply_kernel(const float* __restrict__ T2f, _Float16* __restrict__ T2h,
                               const float* __restrict__ stats,
                               const float* __restrict__ gamma, const float* __restrict__ beta) {
  int i = blockIdx.x * blockDim.x + threadIdx.x;
  const int total = Bb * T2N * NN * CO;
  if (i >= total) return;
  int n = (i >> 6) % NN;
  float v = T2f[i];
  T2h[i] = (_Float16)fmaf((v - stats[n]) * stats[64 + n], gamma[n], beta[n]);
}

// ---------------------------------------------------------------------------
// Kernel 8: conv3 WMMA GEMM: (1088 x 9600) @ (9600 x 128), bias fused
// B = f16 conv3 weights, oc-major -> contiguous per-lane loads
// ---------------------------------------------------------------------------
__global__ void conv3_gemm_kernel(const _Float16* __restrict__ T2h,
                                  const _Float16* __restrict__ Wc, const float* __restrict__ bc,
                                  float* __restrict__ y) {
  int wave = (blockIdx.x * blockDim.x + threadIdx.x) >> 5;
  int lane = threadIdx.x & 31;
  int tn = wave & 7;          // 128/16
  int tm = wave >> 3;         // 68
  int r  = lane & 15;
  int kb = (lane >> 4) << 3;
  int rowA = tm * 16 + r;
  int b_ = rowA / T3;
  int t  = rowA % T3;
  int colB = tn * 16 + r;
  const _Float16* pb0 = Wc + (size_t)colB * 9600;
  v8f acc = {};
  for (int k0 = 0; k0 < 9600; k0 += 32) {
    int kblk = k0 / (NN * CO);        // /3200; chunks never straddle (3200 % 32 == 0)
    int koff = k0 - kblk * (NN * CO);
    v16h a  = frag_load(T2h + (size_t)(b_ * T2N + t + kblk) * (NN * CO) + koff + kb);
    v16h bf = frag_load(pb0 + k0 + kb);
    acc = WMMA(a, bf, acc);
  }
  int n  = lane & 15;
  int mb = (lane >> 4) << 3;
  int oc = tn * 16 + n;
  float bv = bc[oc];
#pragma unroll
  for (int v = 0; v < 8; ++v)
    y[(size_t)(tm * 16 + mb + v) * 128 + oc] = acc[v] + bv;
}

// Kernel 9: mean pool over time
__global__ void pool_kernel(const float* __restrict__ y, float* __restrict__ pooled) {
  int i = blockIdx.x * blockDim.x + threadIdx.x;
  if (i >= Bb * 128) return;
  int b = i >> 7, oc = i & 127;
  float s = 0.0f;
  for (int t = 0; t < T3; ++t) s += y[(size_t)(b * T3 + t) * 128 + oc];
  pooled[i] = s * (1.0f / (float)T3);
}

// Kernel 10: final FC  out[b, j] = pooled[b,:] @ f1_w[:, j] + f1_b[j]
__global__ void fc_kernel(const float* __restrict__ pooled, const float* __restrict__ f1w,
                          const float* __restrict__ f1b, float* __restrict__ out) {
  int i = blockIdx.x * blockDim.x + threadIdx.x;
  const int total = Bb * NN * CO;     // 102400
  if (i >= total) return;
  int b = i / (NN * CO);
  int j = i % (NN * CO);
  float s = f1b[j];
#pragma unroll 8
  for (int oc = 0; oc < 128; ++oc)
    s = fmaf(pooled[b * 128 + oc], f1w[(size_t)oc * (NN * CO) + j], s);
  out[i] = s;
}

// ---------------------------------------------------------------------------
extern "C" void kernel_launch(void* const* d_in, const int* in_sizes, int n_in,
                              void* d_out, int out_size, void* d_ws, size_t ws_size,
                              hipStream_t stream) {
  const float* X       = (const float*)d_in[0];
  const int*   ei      = (const int*)  d_in[1];
  const float* tc1_w1  = (const float*)d_in[2];
  const float* tc1_b1  = (const float*)d_in[3];
  const float* tc1_w2  = (const float*)d_in[4];
  const float* tc1_b2  = (const float*)d_in[5];
  const float* tc1_w3  = (const float*)d_in[6];
  const float* tc1_b3  = (const float*)d_in[7];
  const float* cheb_w  = (const float*)d_in[8];
  const float* cheb_b  = (const float*)d_in[9];
  const float* tc2_w1  = (const float*)d_in[10];
  const float* tc2_b1  = (const float*)d_in[11];
  const float* tc2_w2  = (const float*)d_in[12];
  const float* tc2_b2  = (const float*)d_in[13];
  const float* tc2_w3  = (const float*)d_in[14];
  const float* tc2_b3  = (const float*)d_in[15];
  const float* bn_g    = (const float*)d_in[16];
  const float* bn_b    = (const float*)d_in[17];
  const float* conv3_w = (const float*)d_in[18];
  const float* conv3_b = (const float*)d_in[19];
  const float* f1_w    = (const float*)d_in[20];
  const float* f1_b    = (const float*)d_in[21];
  float* out = (float*)d_out;

  // ---- byte allocator on workspace (256B aligned blocks) ----
  char* base = (char*)d_ws;
  size_t off = 0;
  auto alloc = [&](size_t bytes) -> void* {
    void* p = base + off;
    off += (bytes + 255) & ~(size_t)255;
    return p;
  };
  _Float16* Txh   = (_Float16*)alloc(3 * TXS * sizeof(_Float16)); // T0h/Tx1h/Tx2h contiguous
  _Float16* T0h   = Txh;
  _Float16* Tx1h  = Txh + TXS;
  _Float16* Tx2h  = Txh + 2 * TXS;
  _Float16* Gh    = (_Float16*)alloc(TXS * sizeof(_Float16));
  float*    wE    = (float*)   alloc((size_t)S1 * EE * sizeof(float));
  float*    T2f   = (float*)   alloc((size_t)Bb * T2N * NN * CO * sizeof(float));
  _Float16* T2h   = (_Float16*)alloc((size_t)Bb * T2N * NN * CO * sizeof(_Float16));
  float*    st    = (float*)   alloc(128 * sizeof(float));
  float*    y     = (float*)   alloc((size_t)Bb * T3 * 128 * sizeof(float));
  float*    pl    = (float*)   alloc((size_t)Bb * 128 * sizeof(float));
  _Float16* chebWt= (_Float16*)alloc((size_t)384 * 128 * sizeof(_Float16));
  _Float16* w1t   = (_Float16*)alloc((size_t)384 * CO * sizeof(_Float16));
  _Float16* w2t   = (_Float16*)alloc((size_t)384 * CO * sizeof(_Float16));
  _Float16* w3t   = (_Float16*)alloc((size_t)384 * CO * sizeof(_Float16));
  _Float16* Wc3h  = (_Float16*)alloc((size_t)128 * 9600 * sizeof(_Float16));

  // ---- weight prep (tiny) ----
  transpose_f16_kernel<<<(384 * 128 + 255) / 256, 256, 0, stream>>>(cheb_w, chebWt, 384, 128);
  transpose_f16_kernel<<<(384 * CO + 255) / 256, 256, 0, stream>>>(tc2_w1, w1t, 384, CO);
  transpose_f16_kernel<<<(384 * CO + 255) / 256, 256, 0, stream>>>(tc2_w2, w2t, 384, CO);
  transpose_f16_kernel<<<(384 * CO + 255) / 256, 256, 0, stream>>>(tc2_w3, w3t, 384, CO);
  convert_f16_kernel<<<(128 * 9600 + 255) / 256, 256, 0, stream>>>(conv3_w, Wc3h, 128 * 9600);

  // 1) temporal conv 1 -> T0h (f16)
  {
    int total = Bb * T1 * NN * HH;        // 7,782,400
    tconv1_kernel<<<(total + 255) / 256, 256, 0, stream>>>(
        X, tc1_w1, tc1_b1, tc1_w2, tc1_b2, tc1_w3, tc1_b3, T0h);
  }
  // 2) edge weights
  edgew_kernel<<<S1, 256, 0, stream>>>(ei, wE);
  // 3) Chebyshev propagation: Tx1 = L T0 ; Tx2 = 2 L Tx1 - T0
  prop_kernel<<<S1, 256, 0, stream>>>(T0h, T0h, wE, ei, 1.0f,  0.0f, Tx1h);
  prop_kernel<<<S1, 256, 0, stream>>>(Tx1h, T0h, wE, ei, 2.0f, -1.0f, Tx2h);
  // 4) cheb einsum + bias + relu  (3800 M-tiles x 8 N-tiles, 8 waves/block)
  cheb_gemm_kernel<<<3800, 256, 0, stream>>>(Txh, chebWt, cheb_b, Gh);
  // 5) temporal conv 2 (GLU)      (3600 M-tiles x 4 N-tiles)
  tconv2_gemm_kernel<<<1800, 256, 0, stream>>>(
      Gh, w1t, tc2_b1, w2t, tc2_b2, w3t, tc2_b3, T2f);
  // 6) batch-norm over nodes
  bnstats_kernel<<<NN, 256, 0, stream>>>(T2f, st);
  {
    int total = Bb * T2N * NN * CO;       // 3,686,400
    bnapply_kernel<<<(total + 255) / 256, 256, 0, stream>>>(T2f, T2h, st, bn_g, bn_b);
  }
  // 7) conv3 (68 M-tiles x 8 N-tiles)
  conv3_gemm_kernel<<<68, 256, 0, stream>>>(T2h, Wc3h, conv3_b, y);
  // 8) mean pool + final FC
  pool_kernel<<<(Bb * 128 + 255) / 256, 256, 0, stream>>>(y, pl);
  fc_kernel<<<(Bb * NN * CO + 255) / 256, 256, 0, stream>>>(pl, f1_w, f1_b, out);
}
